// GRU_24910810317336
// MI455X (gfx1250) — matrix operand order repaired and checked by reading, not compiled, not run on hardware
//
#include <hip/hip_runtime.h>
#include <hip/hip_bf16.h>

#define T_STEPS  512
#define BATCH    64
#define IN0      512
#define HID      1024
#define NTILES   256          // (B/16) * (H/16) = 4 * 64
#define NBLOCKS  64           // scan grid: 64 blocks * 4 waves = 256 tile-waves
#define NTHREADS 128
#define PRE_STRIDE 24         // 3 gates * 8 values per lane

typedef __bf16 v16bf __attribute__((ext_vector_type(16)));
typedef float  v8f   __attribute__((ext_vector_type(8)));

union AFrag  { v16bf v; uint4 q[2]; __hip_bfloat16 h[16]; };
union CFrag8 { uint4 q; __hip_bfloat16 h[8]; };

__device__ __forceinline__ float sigmoidf_(float x) {
    return 1.0f / (1.0f + __expf(-x));
}

// ---------------------------------------------------------------------------
// Triple-gate GEMM over K slices of 32, bf16 A source (row-major activations).
// A lane L: row M = L%16, K = {khalf*8..+7, 16+khalf*8..+7}  -> 2 x 16B loads
// B pre-packed: 16 contiguous bf16 per (hj, slice, lane)     -> 1 x 32B load
// ---------------------------------------------------------------------------
__device__ __forceinline__ void mma3_bf16(
    const __hip_bfloat16* __restrict__ Asrc, int lda,
    const __hip_bfloat16* __restrict__ Pr,
    const __hip_bfloat16* __restrict__ Pz,
    const __hip_bfloat16* __restrict__ Pn,
    int bi, int hj, int fin, int lane,
    v8f& cr, v8f& cz, v8f& cn)
{
    const int S     = fin >> 5;
    const int row   = bi * 16 + (lane & 15);
    const int khalf = lane >> 4;
    const __hip_bfloat16* arow = Asrc + (size_t)row * lda + khalf * 8;
    const size_t pbase = ((size_t)hj * S * 32 + lane) * 16;

    for (int s = 0; s < S; ++s) {
        AFrag a;
        a.q[0] = *reinterpret_cast<const uint4*>(arow + s * 32);
        a.q[1] = *reinterpret_cast<const uint4*>(arow + s * 32 + 16);
        const size_t po = pbase + (size_t)s * 512;
        v16bf br = *reinterpret_cast<const v16bf*>(Pr + po);
        v16bf bz = *reinterpret_cast<const v16bf*>(Pz + po);
        v16bf bn = *reinterpret_cast<const v16bf*>(Pn + po);
        cr = __builtin_amdgcn_wmma_f32_16x16x32_bf16(false, a.v, false, br,
                                                     (short)0, cr, false, false);
        cz = __builtin_amdgcn_wmma_f32_16x16x32_bf16(false, a.v, false, bz,
                                                     (short)0, cz, false, false);
        cn = __builtin_amdgcn_wmma_f32_16x16x32_bf16(false, a.v, false, bn,
                                                     (short)0, cn, false, false);
    }
}

// Same, but A source is f32 (converted on the fly; used by layer-0 precompute)
__device__ __forceinline__ void mma3_f32(
    const float* __restrict__ Asrc, int lda,
    const __hip_bfloat16* __restrict__ Pr,
    const __hip_bfloat16* __restrict__ Pz,
    const __hip_bfloat16* __restrict__ Pn,
    int bi, int hj, int fin, int lane,
    v8f& cr, v8f& cz, v8f& cn)
{
    const int S     = fin >> 5;
    const int row   = bi * 16 + (lane & 15);
    const int khalf = lane >> 4;
    const float* arow = Asrc + (size_t)row * lda + khalf * 8;
    const size_t pbase = ((size_t)hj * S * 32 + lane) * 16;

    for (int s = 0; s < S; ++s) {
        const float* p0 = arow + s * 32;
        float4 f0 = *reinterpret_cast<const float4*>(p0);
        float4 f1 = *reinterpret_cast<const float4*>(p0 + 4);
        float4 f2 = *reinterpret_cast<const float4*>(p0 + 16);
        float4 f3 = *reinterpret_cast<const float4*>(p0 + 20);
        AFrag a;
        a.h[0]  = __float2bfloat16(f0.x); a.h[1]  = __float2bfloat16(f0.y);
        a.h[2]  = __float2bfloat16(f0.z); a.h[3]  = __float2bfloat16(f0.w);
        a.h[4]  = __float2bfloat16(f1.x); a.h[5]  = __float2bfloat16(f1.y);
        a.h[6]  = __float2bfloat16(f1.z); a.h[7]  = __float2bfloat16(f1.w);
        a.h[8]  = __float2bfloat16(f2.x); a.h[9]  = __float2bfloat16(f2.y);
        a.h[10] = __float2bfloat16(f2.z); a.h[11] = __float2bfloat16(f2.w);
        a.h[12] = __float2bfloat16(f3.x); a.h[13] = __float2bfloat16(f3.y);
        a.h[14] = __float2bfloat16(f3.z); a.h[15] = __float2bfloat16(f3.w);
        const size_t po = pbase + (size_t)s * 512;
        v16bf br = *reinterpret_cast<const v16bf*>(Pr + po);
        v16bf bz = *reinterpret_cast<const v16bf*>(Pz + po);
        v16bf bn = *reinterpret_cast<const v16bf*>(Pn + po);
        cr = __builtin_amdgcn_wmma_f32_16x16x32_bf16(false, a.v, false, br,
                                                     (short)0, cr, false, false);
        cz = __builtin_amdgcn_wmma_f32_16x16x32_bf16(false, a.v, false, bz,
                                                     (short)0, cz, false, false);
        cn = __builtin_amdgcn_wmma_f32_16x16x32_bf16(false, a.v, false, bn,
                                                     (short)0, cn, false, false);
    }
}

// ---------------------------------------------------------------------------
// Device-wide barrier across NBLOCKS blocks (expected-generation protocol)
// ---------------------------------------------------------------------------
__device__ __forceinline__ void grid_sync(unsigned* cnt, unsigned* gen,
                                          unsigned expected)
{
    __syncthreads();
    if (threadIdx.x == 0) {
        unsigned prev = __hip_atomic_fetch_add(cnt, 1u, __ATOMIC_ACQ_REL,
                                               __HIP_MEMORY_SCOPE_AGENT);
        if (prev == (unsigned)(NBLOCKS - 1)) {
            __hip_atomic_store(cnt, 0u, __ATOMIC_RELAXED,
                               __HIP_MEMORY_SCOPE_AGENT);
            __hip_atomic_store(gen, expected, __ATOMIC_RELEASE,
                               __HIP_MEMORY_SCOPE_AGENT);
        } else {
            while (__hip_atomic_load(gen, __ATOMIC_ACQUIRE,
                                     __HIP_MEMORY_SCOPE_AGENT) < expected) {
                __builtin_amdgcn_s_sleep(1);
            }
        }
    }
    __syncthreads();
}

// ---------------------------------------------------------------------------
// Precompute kernel: pregate[t][tile][lane][24] =
//   bf16( x_t @ {irW,izW,inW}^T  + folded biases ), stored in C-fragment order
// One wave per (t, tile). Grid: T*NTILES/4 blocks of 128 threads.
// ---------------------------------------------------------------------------
struct PreArgs {
    const float*          src_f32;   // layer0: x [T,B,IN0] f32 (else null)
    const __hip_bfloat16* src_bf16;  // layer1: y0 [T,B,H] bf16 (else null)
    int fin;
    const __hip_bfloat16* pki[3];    // packed irW izW inW
    const float* irB; const float* izB; const float* inB;
    const float* hrB; const float* hzB;
    __hip_bfloat16* pre;             // [T][NTILES][32][24]
};

__launch_bounds__(128)
__global__ void pregate_kernel(PreArgs P)
{
    const int lane = threadIdx.x & 31;
    const int w    = blockIdx.x * 4 + (threadIdx.x >> 5);
    if (w >= T_STEPS * NTILES) return;
    const int t    = w >> 8;
    const int tile = w & 255;
    const int bi   = tile & 3;
    const int hj   = tile >> 2;
    const int hcol = hj * 16 + (lane & 15);

    const float br0 = P.irB[hcol] + P.hrB[hcol];
    const float bz0 = P.izB[hcol] + P.hzB[hcol];
    const float bn0 = P.inB[hcol];
    v8f cr, cz, cn;
#pragma unroll
    for (int v = 0; v < 8; ++v) { cr[v] = br0; cz[v] = bz0; cn[v] = bn0; }

    if (P.src_f32) {
        mma3_f32(P.src_f32 + (size_t)t * BATCH * P.fin, P.fin,
                 P.pki[0], P.pki[1], P.pki[2], bi, hj, P.fin, lane, cr, cz, cn);
    } else {
        mma3_bf16(P.src_bf16 + (size_t)t * BATCH * P.fin, P.fin,
                  P.pki[0], P.pki[1], P.pki[2], bi, hj, P.fin, lane, cr, cz, cn);
    }

    __hip_bfloat16* dst = P.pre + ((size_t)w * 32 + lane) * PRE_STRIDE;
    CFrag8 qr, qz, qn;
#pragma unroll
    for (int v = 0; v < 8; ++v) {
        qr.h[v] = __float2bfloat16(cr[v]);
        qz.h[v] = __float2bfloat16(cz[v]);
        qn.h[v] = __float2bfloat16(cn[v]);
    }
    *reinterpret_cast<uint4*>(dst)      = qr.q;
    *reinterpret_cast<uint4*>(dst + 8)  = qz.q;
    *reinterpret_cast<uint4*>(dst + 16) = qn.q;
}

// ---------------------------------------------------------------------------
// Persistent scan kernel: per timestep only the recurrent h-GEMM + pointwise.
// hf_*: f32 hidden state (read t-1 / write t);  hb_*: bf16 shadow for WMMA.
// ---------------------------------------------------------------------------
struct ScanArgsV2 {
    const __hip_bfloat16* pre;       // [T][NTILES][32][24]
    const __hip_bfloat16* pkh[3];    // packed hrW hzW hnW
    const float* hnB;
    float*          hf_base; size_t hf_stride; unsigned hf_mask;
    __hip_bfloat16* hb_base; size_t hb_stride; unsigned hb_mask;
    float* hT_out;                   // final hidden state destination
    unsigned* cnt; unsigned* gen;
};

__launch_bounds__(NTHREADS, 1)
__global__ void scan_kernel(ScanArgsV2 A)
{
    const int lane  = threadIdx.x & 31;
    const int wave  = threadIdx.x >> 5;
    const int tile  = blockIdx.x * (NTHREADS / 32) + wave;  // 0..255
    const int bi    = tile & 3;
    const int hj    = tile >> 2;
    const int khalf = lane >> 4;
    const int hcol  = hj * 16 + (lane & 15);
    const float hnBv = A.hnB[hcol];

    for (int t = 0; t < T_STEPS; ++t) {
        // folded input-projection fragment: 3 x 16B aligned loads
        const __hip_bfloat16* pf =
            A.pre + (((size_t)t * NTILES + tile) * 32 + lane) * PRE_STRIDE;
        CFrag8 fr, fz, fn;
        fr.q = *reinterpret_cast<const uint4*>(pf);
        fz.q = *reinterpret_cast<const uint4*>(pf + 8);
        fn.q = *reinterpret_cast<const uint4*>(pf + 16);

        v8f cr = {0.f,0.f,0.f,0.f,0.f,0.f,0.f,0.f};
        v8f cz = {0.f,0.f,0.f,0.f,0.f,0.f,0.f,0.f};
        v8f cn;
#pragma unroll
        for (int v = 0; v < 8; ++v) cn[v] = hnBv;   // hnB folded into C init

        if (t > 0) {
            const __hip_bfloat16* hb =
                A.hb_base + (size_t)((t - 1) & A.hb_mask) * A.hb_stride;
            mma3_bf16(hb, HID, A.pkh[0], A.pkh[1], A.pkh[2],
                      bi, hj, HID, lane, cr, cz, cn);
        }

        const float* hfp =
            A.hf_base + (size_t)((t - 1) & A.hf_mask) * A.hf_stride;
        float* hfo = A.hf_base + (size_t)(t & A.hf_mask) * A.hf_stride;
        __hip_bfloat16* hbo =
            A.hb_base + (size_t)(t & A.hb_mask) * A.hb_stride;

#pragma unroll
        for (int v = 0; v < 8; ++v) {
            const int brow = bi * 16 + v + 8 * khalf;
            const size_t idx = (size_t)brow * HID + hcol;
            float r  = sigmoidf_(__bfloat162float(fr.h[v]) + cr[v]);
            float z  = sigmoidf_(__bfloat162float(fz.h[v]) + cz[v]);
            float nn = tanhf(__bfloat162float(fn.h[v]) + r * cn[v]);
            float hp = (t > 0) ? hfp[idx] : 0.0f;
            float hnew = (1.0f - z) * nn + z * hp;
            hfo[idx] = hnew;
            hbo[idx] = __float2bfloat16(hnew);
        }
        grid_sync(A.cnt, A.gen, (unsigned)(t + 1));
    }

    // final hidden state (all writes fenced by the last grid_sync)
    const float* src =
        A.hf_base + (size_t)((T_STEPS - 1) & A.hf_mask) * A.hf_stride;
    for (int i = blockIdx.x * NTHREADS + threadIdx.x; i < BATCH * HID;
         i += NBLOCKS * NTHREADS)
        A.hT_out[i] = src[i];
}

// ---------------------------------------------------------------------------
// Weight packing into per-lane WMMA B-fragment layout:
//   P[((hj*S + s)*32 + lane)*16 + e] =
//     W[(hj*16 + lane%16)*fin + s*32 + (lane/16)*8 + (e<8 ? e : e+8)]
// ---------------------------------------------------------------------------
__global__ void pack_w_kernel(const float* __restrict__ W,
                              __hip_bfloat16* __restrict__ P, int fin)
{
    const int S = fin >> 5;
    const int total = (HID / 16) * S * 32;
    int g = blockIdx.x * blockDim.x + threadIdx.x;
    if (g >= total) return;
    const int lane = g & 31;
    const int s    = (g >> 5) % S;
    const int hj   = (g >> 5) / S;
    const float* src = W + (size_t)(hj * 16 + (lane & 15)) * fin
                         + s * 32 + (lane >> 4) * 8;
    __hip_bfloat16* dst = P + (size_t)g * 16;
#pragma unroll
    for (int e = 0; e < 16; ++e) dst[e] = __float2bfloat16(src[e < 8 ? e : e + 8]);
}

// ---------------------------------------------------------------------------
extern "C" void kernel_launch(void* const* d_in, const int* in_sizes, int n_in,
                              void* d_out, int out_size, void* d_ws,
                              size_t ws_size, hipStream_t stream)
{
    (void)in_sizes; (void)n_in; (void)out_size; (void)ws_size;

    const float* x = (const float*)d_in[0];
    const float* W[12]; const float* Bz[12];
    for (int i = 0; i < 6; ++i) W[i]      = (const float*)d_in[1 + i];
    for (int i = 0; i < 6; ++i) Bz[i]     = (const float*)d_in[7 + i];
    for (int i = 0; i < 6; ++i) W[6 + i]  = (const float*)d_in[13 + i];
    for (int i = 0; i < 6; ++i) Bz[6 + i] = (const float*)d_in[19 + i];

    char* ws = (char*)d_ws;
    size_t off = 0;
    auto alloc = [&](size_t bytes) {
        size_t o = off;
        off += (bytes + 255) & ~(size_t)255;
        return o;
    };

    const size_t pre_off = alloc((size_t)T_STEPS * NTILES * 32 * PRE_STRIDE * 2);
    const size_t y0_off  = alloc((size_t)T_STEPS * BATCH * HID * 2);
    const size_t h0f_off[2] = { alloc((size_t)BATCH * HID * 4),
                                alloc((size_t)BATCH * HID * 4) };
    const size_t h1b_off[2] = { alloc((size_t)BATCH * HID * 2),
                                alloc((size_t)BATCH * HID * 2) };
    static const int fins[12] = { IN0, IN0, IN0, HID, HID, HID,
                                  HID, HID, HID, HID, HID, HID };
    size_t pk_off[12];
    for (int i = 0; i < 12; ++i) pk_off[i] = alloc((size_t)HID * fins[i] * 2);
    const size_t bar_off = alloc(256);

    hipMemsetAsync(ws + bar_off, 0, 256, stream);

    // pack all 12 weight matrices to bf16 fragment layout
    for (int i = 0; i < 12; ++i) {
        const int total = (HID / 16) * (fins[i] >> 5) * 32;
        pack_w_kernel<<<(total + 255) / 256, 256, 0, stream>>>(
            W[i], (__hip_bfloat16*)(ws + pk_off[i]), fins[i]);
    }

    __hip_bfloat16* pre = (__hip_bfloat16*)(ws + pre_off);
    __hip_bfloat16* y0  = (__hip_bfloat16*)(ws + y0_off);
    float* out = (float*)d_out;
    const int pre_blocks = T_STEPS * NTILES / 4;

    // ---------------- layer 0 ----------------
    {
        PreArgs P;
        P.src_f32 = x; P.src_bf16 = nullptr; P.fin = IN0;
        for (int i = 0; i < 3; ++i) P.pki[i] = (const __hip_bfloat16*)(ws + pk_off[i]);
        P.irB = Bz[0]; P.izB = Bz[1]; P.inB = Bz[2];
        P.hrB = Bz[3]; P.hzB = Bz[4];
        P.pre = pre;
        pregate_kernel<<<pre_blocks, 128, 0, stream>>>(P);

        ScanArgsV2 A;
        A.pre = pre;
        for (int i = 0; i < 3; ++i) A.pkh[i] = (const __hip_bfloat16*)(ws + pk_off[3 + i]);
        A.hnB = Bz[5];
        A.hf_base = (float*)(ws + h0f_off[0]);          // ping-pong (contiguous pair)
        A.hf_stride = (size_t)BATCH * HID; A.hf_mask = 1u;
        A.hb_base = y0;                                  // y0 slices double as bf16 h
        A.hb_stride = (size_t)BATCH * HID; A.hb_mask = (unsigned)(T_STEPS - 1);
        A.hT_out = out + (size_t)T_STEPS * BATCH * HID;  // hT0
        A.cnt = (unsigned*)(ws + bar_off);
        A.gen = (unsigned*)(ws + bar_off + 8);
        scan_kernel<<<NBLOCKS, NTHREADS, 0, stream>>>(A);
    }

    // ---------------- layer 1 ----------------
    {
        PreArgs P;
        P.src_f32 = nullptr; P.src_bf16 = y0; P.fin = HID;
        for (int i = 0; i < 3; ++i) P.pki[i] = (const __hip_bfloat16*)(ws + pk_off[6 + i]);
        P.irB = Bz[6]; P.izB = Bz[7]; P.inB = Bz[8];
        P.hrB = Bz[9]; P.hzB = Bz[10];
        P.pre = pre;                                     // reuse buffer
        pregate_kernel<<<pre_blocks, 128, 0, stream>>>(P);

        ScanArgsV2 A;
        A.pre = pre;
        for (int i = 0; i < 3; ++i) A.pkh[i] = (const __hip_bfloat16*)(ws + pk_off[9 + i]);
        A.hnB = Bz[11];
        A.hf_base = out;                                 // y1 slices hold f32 h
        A.hf_stride = (size_t)BATCH * HID; A.hf_mask = (unsigned)(T_STEPS - 1);
        A.hb_base = (__hip_bfloat16*)(ws + h1b_off[0]);  // bf16 ping-pong
        A.hb_stride = (size_t)BATCH * HID; A.hb_mask = 1u;
        A.hT_out = out + (size_t)T_STEPS * BATCH * HID + (size_t)BATCH * HID; // hT1
        A.cnt = (unsigned*)(ws + bar_off + 16);
        A.gen = (unsigned*)(ws + bar_off + 24);
        scan_kernel<<<NBLOCKS, NTHREADS, 0, stream>>>(A);
    }
}
